// MultiHeadSelfAttention_86741159510022
// MI455X (gfx1250) — matrix operand order repaired and checked by reading, not compiled, not run on hardware
//
#include <hip/hip_runtime.h>
#include <hip/hip_bf16.h>

// Problem constants (match reference)
#define DMODEL 1024
#define NHEADS 16
#define HDIM   64
#define BATCH  2
#define SEQ    2048
#define MTOT   (BATCH * SEQ)   // 4096 rows in all big GEMMs

typedef __attribute__((ext_vector_type(16))) _Float16 v16h;
typedef __attribute__((ext_vector_type(8)))  _Float16 v8h;
typedef __attribute__((ext_vector_type(4)))  _Float16 v4h;
typedef __attribute__((ext_vector_type(8)))  float    v8f;
typedef __attribute__((ext_vector_type(4)))  int      v4i;

// ---------------------------------------------------------------------------
// Async global->LDS copy (CDNA5 GLOBAL_LOAD_ASYNC_TO_LDS_B128, ASYNCcnt) with
// a compile-time fallback to a plain LDS store if the builtin is absent.
// Probe-discovered signature: (v4i addrspace(1)*, v4i addrspace(3)*, i32, i32)
// ---------------------------------------------------------------------------
#if defined(__has_builtin)
#  if __has_builtin(__builtin_amdgcn_global_load_async_to_lds_b128)
#    define USE_ASYNC_LDS 1
#  endif
#endif
#ifndef USE_ASYNC_LDS
#  define USE_ASYNC_LDS 0
#endif

typedef __attribute__((address_space(1))) v4i* gbl_v4i_ptr;
typedef __attribute__((address_space(3))) v4i* lds_v4i_ptr;

static __device__ __forceinline__ void cp_b128(void* lds, const _Float16* g) {
#if USE_ASYNC_LDS
    __builtin_amdgcn_global_load_async_to_lds_b128(
        (gbl_v4i_ptr)(v4i*)g,
        (lds_v4i_ptr)(v4i*)lds,
        /*offset=*/0, /*cpol=*/0);
#else
    *(v8h*)lds = *(const v8h*)g;
#endif
}

static __device__ __forceinline__ void wait_async_copies() {
#if USE_ASYNC_LDS
#  if defined(__has_builtin) && __has_builtin(__builtin_amdgcn_s_wait_asynccnt)
    __builtin_amdgcn_s_wait_asynccnt(0);
#  else
    asm volatile("s_wait_asynccnt 0" ::: "memory");
#  endif
#endif
}

static __device__ __forceinline__ v8f zero8() {
    v8f z = {0.f, 0.f, 0.f, 0.f, 0.f, 0.f, 0.f, 0.f};
    return z;
}

// ---------------------------------------------------------------------------
// WMMA wrapper: D = A(16x32 f16) * B(32x16 f16) + C(16x16 f32)
// ---------------------------------------------------------------------------
static __device__ __forceinline__ v8f wmma_f16(v16h a, v16h b, v8f c) {
    return __builtin_amdgcn_wmma_f32_16x16x32_f16(
        /*neg_a=*/false, a, /*neg_b=*/false, b,
        /*c_mod=*/(short)0, c, /*reuse_a=*/false, /*reuse_b=*/false);
}

// ---------------------------------------------------------------------------
// A-operand (16x32, MxK) load from row-major [M x ld] f16 buffer.
// ISA layout: lane l -> row l%16; element e -> K = 16*(e/8) + 8*(l/16) + e%8
// => two contiguous 16-byte chunks per lane at k-offsets 8h and 16+8h.
// Works for both global and LDS source pointers (inlined, AS inferred).
// ---------------------------------------------------------------------------
static __device__ __forceinline__ v16h load_frag_a(const _Float16* __restrict__ p, int ld) {
    const int lane = threadIdx.x & 31;
    const int r  = lane & 15;
    const int hh = lane >> 4;
    const _Float16* q = p + (size_t)r * ld + 8 * hh;
    v8h lo = *(const v8h*)(q);
    v8h hi = *(const v8h*)(q + 16);
    v16h out;
#pragma unroll
    for (int i = 0; i < 8; ++i) { out[i] = lo[i]; out[i + 8] = hi[i]; }
    return out;
}

// ---------------------------------------------------------------------------
// B-operand (32x16, KxN) load from a buffer laid out as [N x ld] row-major
// with K contiguous in each row (i.e. B[k][n] = buf[n*ld + k]).
// ISA layout: lane l -> col l%16; element e -> K = 16*(e/8) + 8*(l/16) + e%8
// => two 16-byte chunks per lane at k-offsets 8h and 16+8h (matches A).
// ---------------------------------------------------------------------------
static __device__ __forceinline__ v16h load_frag_b(const _Float16* __restrict__ p, int ld) {
    const int lane = threadIdx.x & 31;
    const _Float16* q = p + (size_t)(lane & 15) * ld + 8 * (lane >> 4);
    v8h lo = *(const v8h*)(q);
    v8h hi = *(const v8h*)(q + 16);
    v16h out;
#pragma unroll
    for (int i = 0; i < 8; ++i) { out[i] = lo[i]; out[i + 8] = hi[i]; }
    return out;
}

// ---------------------------------------------------------------------------
// fp32 -> f16 conversion (vectorized x4)
// ---------------------------------------------------------------------------
__global__ void cvt_f32_to_f16_kernel(const float* __restrict__ in,
                                      _Float16* __restrict__ out, int n4) {
    int i = blockIdx.x * blockDim.x + threadIdx.x;
    if (i < n4) {
        float4 v = ((const float4*)in)[i];
        v4h o;
        o[0] = (_Float16)v.x; o[1] = (_Float16)v.y;
        o[2] = (_Float16)v.z; o[3] = (_Float16)v.w;
        ((v4h*)out)[i] = o;
    }
}

// ---------------------------------------------------------------------------
// Generic WMMA GEMM: C[M x N] = A[M x K] * W[N x K]^T  (i.e. out = A @ W.T)
// M = 4096, N = K = 1024. Block = 8 waves (2x4), block tile 64x256x32,
// wave tile 32x64. A/B panels double-buffered in LDS via async copies.
// LDS rows padded to 40 halves (80 B): 16 consecutive rows hit 16 distinct
// bank groups (gcd(20,64)=4 -> period 16) => conflict-free fragment reads.
// MODE 0: store f16 into (B, H, S, HDIM)    -- Q and K
// MODE 1: store f16 into (B, H, HDIM, S)    -- transposed V (packed b128 stores)
// MODE 2: store f32 row-major (B*S, DMODEL) -- final output
// ---------------------------------------------------------------------------
template <int MODE>
__global__ void __launch_bounds__(256)
gemm_wmma_kernel(const _Float16* __restrict__ A, const _Float16* __restrict__ W,
                 _Float16* __restrict__ outh, float* __restrict__ outf) {
    __shared__ __align__(16) _Float16 Alds[2][64][40];
    __shared__ __align__(16) _Float16 Blds[2][256][40];

    const int K = DMODEL;
    const int lane = threadIdx.x & 31;
    const int wave = threadIdx.x >> 5;
    const int wm = wave >> 2;            // 0..1
    const int wn = wave & 3;             // 0..3
    const int m0 = blockIdx.x * 64 + wm * 32;
    const int n0 = blockIdx.y * 256 + wn * 64;
    const int m0blk = blockIdx.x * 64;
    const int n0blk = blockIdx.y * 256;

    const int trow = threadIdx.x >> 2;   // 0..63
    const int tseg = (threadIdx.x & 3) * 8;

    v8f acc[2][4];
#pragma unroll
    for (int i = 0; i < 2; ++i)
#pragma unroll
        for (int j = 0; j < 4; ++j) acc[i][j] = zero8();

    // Prologue: stage k-tile 0 into buffer 0
    cp_b128(&Alds[0][trow][tseg], A + (size_t)(m0blk + trow) * K + tseg);
#pragma unroll
    for (int u = 0; u < 4; ++u)
        cp_b128(&Blds[0][trow + u * 64][tseg],
                W + (size_t)(n0blk + trow + u * 64) * K + tseg);

    int buf = 0;
    for (int k0 = 0; k0 < K; k0 += 32) {
        wait_async_copies();         // my copies into `buf` are complete
        __syncthreads();             // everyone's copies into `buf` are complete

        if (k0 + 32 < K) {           // stage next k-tile into the other buffer
            const int kn = k0 + 32;
            __builtin_prefetch(A + (size_t)(m0blk + trow) * K + kn + 32, 0, 3);
            __builtin_prefetch(W + (size_t)(n0blk + trow) * K + kn + 32, 0, 3);
            cp_b128(&Alds[buf ^ 1][trow][tseg],
                    A + (size_t)(m0blk + trow) * K + kn + tseg);
#pragma unroll
            for (int u = 0; u < 4; ++u)
                cp_b128(&Blds[buf ^ 1][trow + u * 64][tseg],
                        W + (size_t)(n0blk + trow + u * 64) * K + kn + tseg);
        }

        // WMMA operands straight from LDS (ds_load_b128 pairs)
        v16h a0 = load_frag_a(&Alds[buf][wm * 32 +  0][0], 40);
        v16h a1 = load_frag_a(&Alds[buf][wm * 32 + 16][0], 40);
        v16h b0 = load_frag_b(&Blds[buf][wn * 64 +  0][0], 40);
        v16h b1 = load_frag_b(&Blds[buf][wn * 64 + 16][0], 40);
        v16h b2 = load_frag_b(&Blds[buf][wn * 64 + 32][0], 40);
        v16h b3 = load_frag_b(&Blds[buf][wn * 64 + 48][0], 40);
        acc[0][0] = wmma_f16(a0, b0, acc[0][0]);
        acc[0][1] = wmma_f16(a0, b1, acc[0][1]);
        acc[0][2] = wmma_f16(a0, b2, acc[0][2]);
        acc[0][3] = wmma_f16(a0, b3, acc[0][3]);
        acc[1][0] = wmma_f16(a1, b0, acc[1][0]);
        acc[1][1] = wmma_f16(a1, b1, acc[1][1]);
        acc[1][2] = wmma_f16(a1, b2, acc[1][2]);
        acc[1][3] = wmma_f16(a1, b3, acc[1][3]);
        buf ^= 1;
    }

    // C layout: lane l -> col n0 + l%16 ; VGPR r -> row m0 + r + 8*(l/16)
    const int hh = lane >> 4;
    const int c  = lane & 15;
    if (MODE == 1) {
        // Transposed-V store: for fixed (i,j), a lane's 8 rows are contiguous
        // along S at fixed head-dim => one 16-byte store per fragment.
#pragma unroll
        for (int i = 0; i < 2; ++i)
#pragma unroll
            for (int j = 0; j < 4; ++j) {
                const int m = m0 + i * 16 + 8 * hh;      // s base (8-aligned)
                const int n = n0 + j * 16 + c;
                const int b  = m >> 11;
                const int s  = m & (SEQ - 1);
                const int hd = n >> 6;
                const int d  = n & (HDIM - 1);
                v8h pk;
#pragma unroll
                for (int r = 0; r < 8; ++r) pk[r] = (_Float16)acc[i][j][r];
                *(v8h*)&outh[(((size_t)(b * NHEADS + hd)) * HDIM + d) * SEQ + s] = pk;
            }
    } else {
#pragma unroll
        for (int i = 0; i < 2; ++i)
#pragma unroll
            for (int j = 0; j < 4; ++j)
#pragma unroll
                for (int r = 0; r < 8; ++r) {
                    const int m = m0 + i * 16 + r + 8 * hh;
                    const int n = n0 + j * 16 + c;
                    const float v = acc[i][j][r];
                    if (MODE == 2) {
                        outf[(size_t)m * DMODEL + n] = v;
                    } else {
                        const int b  = m >> 11;
                        const int s  = m & (SEQ - 1);
                        const int hd = n >> 6;
                        const int d  = n & (HDIM - 1);
                        outh[(((size_t)(b * NHEADS + hd)) * SEQ + s) * HDIM + d] = (_Float16)v;
                    }
                }
    }
}

// ---------------------------------------------------------------------------
// RoPE applied in place to Q and K, layout (B, H, S, HDIM), interleaved pairs.
// ---------------------------------------------------------------------------
__global__ void rope_kernel(_Float16* __restrict__ q, _Float16* __restrict__ k,
                            const int* __restrict__ pos) {
    const int idx = blockIdx.x * blockDim.x + threadIdx.x;
    const int i  = idx & 31;
    const int s  = (idx >> 5) & (SEQ - 1);
    const int bh = idx >> 16;                       // / (32 * SEQ)
    const float p = (float)pos[s];
    // freq = 10000^(-2i/64) = exp(-(i/32) * ln(10000))
    const float freq = __expf(-(float)i * 0.03125f * 9.210340372f);
    const float ang = p * freq;
    float sn, cs;
    __sincosf(ang, &sn, &cs);
    const size_t base = ((size_t)bh * SEQ + s) * HDIM + 2 * i;
    {
        float e = (float)q[base], o = (float)q[base + 1];
        q[base]     = (_Float16)(e * cs - o * sn);
        q[base + 1] = (_Float16)(e * sn + o * cs);
    }
    {
        float e = (float)k[base], o = (float)k[base + 1];
        k[base]     = (_Float16)(e * cs - o * sn);
        k[base + 1] = (_Float16)(e * sn + o * cs);
    }
}

// ---------------------------------------------------------------------------
// Flash attention: grid.y = (b,h), grid.x = S/64. 4 waves/block, each wave
// owns 16 query rows. K/V tiles are staged ONCE per block into LDS with
// async copies (double-buffered), instead of 4x redundant global reads.
// The key-block loop count is block-uniform (2*bx + 2) so the barrier is
// legal; waves past their causal range process fully-masked blocks, which
// contribute exp(-1e30 - mrow) = 0 (mrow is real after block 0, since block
// 0 always contains unmasked keys for every row).
// Online softmax in f32 C-fragments; P routed through LDS to re-layout
// C-fragment -> A-operand for the P*V WMMA.
// Qh,Kh: (B,H,S,HDIM) f16 row-major. Vt: (B,H,HDIM,S) f16. Oh: (B,S,DMODEL) f16.
// ---------------------------------------------------------------------------
__global__ void __launch_bounds__(128)
attn_kernel(const _Float16* __restrict__ Qh, const _Float16* __restrict__ Kh,
            const _Float16* __restrict__ Vt, _Float16* __restrict__ Oh) {
    __shared__ __align__(16) _Float16 Klds[2][32][72];  // 32 keys x 64 d (pad 72)
    __shared__ __align__(16) _Float16 Vlds[2][64][40];  // 64 d x 32 keys (pad 40)
    __shared__ __align__(16) _Float16 plds[4][16][40];  // per-wave P tile

    const int lane = threadIdx.x & 31;
    const int wave = threadIdx.x >> 5;
    const int hh = lane >> 4;
    const int c  = lane & 15;
    const int bh = blockIdx.y;          // b*NHEADS + h
    const int b  = bh >> 4;
    const int hd = bh & (NHEADS - 1);
    const int q0 = blockIdx.x * 64 + wave * 16;
    const int t  = threadIdx.x;

    const _Float16* Qhead = Qh + (size_t)bh * SEQ * HDIM;
    const _Float16* Khead = Kh + (size_t)bh * SEQ * HDIM;
    const _Float16* Vhead = Vt + (size_t)bh * HDIM * SEQ;

    // Q fragments for this wave's 16 rows, pre-scaled by 1/sqrt(HDIM) = 0.125
    v16h qf0 = load_frag_a(Qhead + (size_t)q0 * HDIM + 0,  HDIM);
    v16h qf1 = load_frag_a(Qhead + (size_t)q0 * HDIM + 32, HDIM);
#pragma unroll
    for (int i = 0; i < 16; ++i) {
        qf0[i] = qf0[i] * (_Float16)0.125f;
        qf1[i] = qf1[i] * (_Float16)0.125f;
    }

    v8f o[4];
#pragma unroll
    for (int nt = 0; nt < 4; ++nt) o[nt] = zero8();
    float mrow[8], lrow[8];
#pragma unroll
    for (int r = 0; r < 8; ++r) { mrow[r] = -1e30f; lrow[r] = 0.f; }

    // Stage key-block `kbase` (32 keys) into buffer `bufn`:
    // K: 32 rows x 8 segs (8 halves) = 256 chunks, 2 per thread
    // V: 64 rows x 4 segs            = 256 chunks, 2 per thread
    auto stage_kv = [&](int bufn, int kbase) {
#pragma unroll
        for (int u = 0; u < 2; ++u) {
            const int id  = t + u * 128;
            const int row = id >> 3;                 // 0..31
            const int seg = (id & 7) * 8;            // 0..56
            cp_b128(&Klds[bufn][row][seg],
                    Khead + (size_t)(kbase + row) * HDIM + seg);
        }
#pragma unroll
        for (int u = 0; u < 2; ++u) {
            const int id  = t + u * 128;
            const int row = id >> 2;                 // 0..63
            const int seg = (id & 3) * 8;            // 0..24
            cp_b128(&Vlds[bufn][row][seg],
                    Vhead + (size_t)row * SEQ + kbase + seg);
        }
    };

    const int nkb = 2 * blockIdx.x + 2;   // block-uniform causal trip count
    stage_kv(0, 0);
    int buf = 0;
    for (int kb = 0; kb < nkb; ++kb) {
        wait_async_copies();
        __syncthreads();
        if (kb + 1 < nkb) stage_kv(buf ^ 1, (kb + 1) * 32);

        const int kbase = kb * 32;

        // S tile (16 q-rows x 32 keys) = two 16x16 C-fragments, K from LDS
        v8f s0 = zero8(), s1 = zero8();
        s0 = wmma_f16(qf0, load_frag_b(&Klds[buf][ 0][ 0], 72), s0);
        s0 = wmma_f16(qf1, load_frag_b(&Klds[buf][ 0][32], 72), s0);
        s1 = wmma_f16(qf0, load_frag_b(&Klds[buf][16][ 0], 72), s1);
        s1 = wmma_f16(qf1, load_frag_b(&Klds[buf][16][32], 72), s1);

        const int key0 = kbase + c;
        const int key1 = kbase + 16 + c;
#pragma unroll
        for (int r = 0; r < 8; ++r) {
            const int row = q0 + r + 8 * hh;
            float x0 = (key0 <= row) ? s0[r] : -1e30f;   // causal mask
            float x1 = (key1 <= row) ? s1[r] : -1e30f;
            float mx = fmaxf(x0, x1);
            mx = fmaxf(mx, __shfl_xor(mx, 1));           // reduce over 16 cols
            mx = fmaxf(mx, __shfl_xor(mx, 2));           // (stays in 16-lane half)
            mx = fmaxf(mx, __shfl_xor(mx, 4));
            mx = fmaxf(mx, __shfl_xor(mx, 8));
            const float mnew  = fmaxf(mrow[r], mx);
            const float scale = __expf(mrow[r] - mnew);
            mrow[r] = mnew;
            const float p0 = __expf(x0 - mnew);
            const float p1 = __expf(x1 - mnew);
            float ps = p0 + p1;
            ps += __shfl_xor(ps, 1);
            ps += __shfl_xor(ps, 2);
            ps += __shfl_xor(ps, 4);
            ps += __shfl_xor(ps, 8);
            lrow[r] = lrow[r] * scale + ps;
#pragma unroll
            for (int nt = 0; nt < 4; ++nt) o[nt][r] *= scale;
            // P -> LDS, row-major 16x32 per wave (C-layout scatter)
            plds[wave][r + 8 * hh][c]      = (_Float16)p0;
            plds[wave][r + 8 * hh][c + 16] = (_Float16)p1;
        }

        asm volatile("s_wait_dscnt 0" ::: "memory");

        // Reload P as A-operand (16x32) from LDS
        v16h pf;
        {
            const _Float16* pr = &plds[wave][c][0];
            v8h lo = *(const v8h*)(pr + 8 * hh);
            v8h hi = *(const v8h*)(pr + 16 + 8 * hh);
#pragma unroll
            for (int i = 0; i < 8; ++i) { pf[i] = lo[i]; pf[i + 8] = hi[i]; }
        }

        // O += P * V ; V B-operand from LDS (d-major, keys contiguous)
#pragma unroll
        for (int nt = 0; nt < 4; ++nt)
            o[nt] = wmma_f16(pf, load_frag_b(&Vlds[buf][nt * 16][0], 40), o[nt]);

        buf ^= 1;
    }

    // Normalize and store f16 into (B, S, DMODEL)
#pragma unroll
    for (int r = 0; r < 8; ++r) {
        const int row = q0 + r + 8 * hh;
        const float inv = 1.0f / lrow[r];
#pragma unroll
        for (int nt = 0; nt < 4; ++nt)
            Oh[((size_t)b * SEQ + row) * DMODEL + hd * HDIM + nt * 16 + c] =
                (_Float16)(o[nt][r] * inv);
    }
}

// ---------------------------------------------------------------------------
// Host-side launcher
// ---------------------------------------------------------------------------
extern "C" void kernel_launch(void* const* d_in, const int* in_sizes, int n_in,
                              void* d_out, int out_size, void* d_ws, size_t ws_size,
                              hipStream_t stream) {
    (void)in_sizes; (void)n_in; (void)out_size; (void)ws_size;
    const float* x  = (const float*)d_in[0];
    const int*   tp = (const int*)d_in[1];
    const float* Wq = (const float*)d_in[2];
    const float* Wk = (const float*)d_in[3];
    const float* Wv = (const float*)d_in[4];
    const float* Wo = (const float*)d_in[5];
    float* out = (float*)d_out;

    char* ws = (char*)d_ws;
    _Float16* xh  = (_Float16*)(ws + ((size_t)0  << 20));  // 8 MB
    _Float16* Wqh = (_Float16*)(ws + ((size_t)8  << 20));  // 2 MB each
    _Float16* Wkh = (_Float16*)(ws + ((size_t)10 << 20));
    _Float16* Wvh = (_Float16*)(ws + ((size_t)12 << 20));
    _Float16* Woh = (_Float16*)(ws + ((size_t)14 << 20));
    _Float16* Qh  = (_Float16*)(ws + ((size_t)16 << 20));  // 8 MB (B,H,S,HD)
    _Float16* Kh  = (_Float16*)(ws + ((size_t)24 << 20));  // 8 MB (B,H,S,HD)
    _Float16* Vt  = (_Float16*)(ws + ((size_t)32 << 20));  // 8 MB (B,H,HD,S)
    _Float16* Oh  = (_Float16*)(ws + ((size_t)40 << 20));  // 8 MB (B,S,D)

    // fp32 -> f16 converts
    {
        int n4 = MTOT * DMODEL / 4;
        cvt_f32_to_f16_kernel<<<n4 / 256, 256, 0, stream>>>(x, xh, n4);
        n4 = DMODEL * DMODEL / 4;
        cvt_f32_to_f16_kernel<<<n4 / 256, 256, 0, stream>>>(Wq, Wqh, n4);
        cvt_f32_to_f16_kernel<<<n4 / 256, 256, 0, stream>>>(Wk, Wkh, n4);
        cvt_f32_to_f16_kernel<<<n4 / 256, 256, 0, stream>>>(Wv, Wvh, n4);
        cvt_f32_to_f16_kernel<<<n4 / 256, 256, 0, stream>>>(Wo, Woh, n4);
    }

    // QKV projections
    dim3 gg(MTOT / 64, DMODEL / 256);
    gemm_wmma_kernel<0><<<gg, 256, 0, stream>>>(xh, Wqh, Qh, nullptr);
    gemm_wmma_kernel<0><<<gg, 256, 0, stream>>>(xh, Wkh, Kh, nullptr);
    gemm_wmma_kernel<1><<<gg, 256, 0, stream>>>(xh, Wvh, Vt, nullptr);

    // RoPE on Q and K
    {
        int nthr = BATCH * NHEADS * SEQ * 32;
        rope_kernel<<<nthr / 256, 256, 0, stream>>>(Qh, Kh, tp);
    }

    // Flash attention
    dim3 ga(SEQ / 64, BATCH * NHEADS);
    attn_kernel<<<ga, 128, 0, stream>>>(Qh, Kh, Vt, Oh);

    // Output projection (fp32 result)
    gemm_wmma_kernel<2><<<gg, 256, 0, stream>>>(Oh, Woh, nullptr, out);
}
